// Block_59983513256143
// MI455X (gfx1250) — matrix operand order repaired and checked by reading, not compiled, not run on hardware
//
#include <hip/hip_runtime.h>
#include <hip/hip_bf16.h>

#define DEV_INLINE __device__ __forceinline__

typedef __attribute__((ext_vector_type(16))) __bf16 v16bf;
typedef __attribute__((ext_vector_type(8)))  __bf16 v8bf;
typedef __attribute__((ext_vector_type(8)))  float  v8f;

constexpr int   B_    = 4;
constexpr int   SEQ   = 2048;
constexpr int   C_    = 384;
constexpr int   H_    = 6;
constexpr int   D_    = 64;
constexpr int   HID_  = 1536;
constexpr int   M_TOK = B_ * SEQ;          // 8192 token rows
constexpr float SCALE = 0.125f;            // 64^-0.5
constexpr float EPS_  = 1e-5f;

// ---------------- workspace layout (bytes, all 4 KiB aligned) ----------------
constexpr size_t OFF_WQKV = 0;                                   // [C,3C]  bf16
constexpr size_t OFF_WPROJ = OFF_WQKV + (size_t)C_ * 3 * C_ * 2; // [C,C]   bf16
constexpr size_t OFF_WFC1 = OFF_WPROJ + (size_t)C_ * C_ * 2;     // [C,HID] bf16
constexpr size_t OFF_WFC2 = OFF_WFC1 + (size_t)C_ * HID_ * 2;    // [HID,C] bf16
constexpr size_t OFF_H    = OFF_WFC2 + (size_t)HID_ * C_ * 2;    // [M,C]   bf16 (LN out)
constexpr size_t OFF_Q    = OFF_H   + (size_t)M_TOK * C_ * 2;    // [B,H,N,D] bf16
constexpr size_t OFF_KT   = OFF_Q   + (size_t)M_TOK * C_ * 2;    // [B,H,D,N] bf16
constexpr size_t OFF_V    = OFF_KT  + (size_t)M_TOK * C_ * 2;    // [B,H,N,D] bf16
constexpr size_t OFF_O    = OFF_V   + (size_t)M_TOK * C_ * 2;    // [M,C]   bf16 (attn out)
constexpr size_t OFF_X1   = OFF_O   + (size_t)M_TOK * C_ * 2;    // [M,C]   bf16 (resid1)
constexpr size_t OFF_HID  = OFF_X1  + (size_t)M_TOK * C_ * 2;    // [M,HID] bf16 (gelu out)

// ---------------- WMMA helpers ----------------
DEV_INLINE v8f wmma_bf16(v16bf a, v16bf b, v8f c) {
  // D = A(16x32 bf16) * B(32x16 bf16) + C(16x16 f32)
  return __builtin_amdgcn_wmma_f32_16x16x32_bf16(false, a, false, b, (short)0, c,
                                                 false, false);
}

// A-fragment: lane holds row M = (lane&15); element e -> K = k0 + (lane>>4)*8 + (e<8?e:e+8)
DEV_INLINE v16bf load_a16x32(const __bf16* __restrict__ base, int ld, int row,
                             int k0, int lane) {
  const __bf16* p = base + (size_t)row * ld + k0 + ((lane >> 4) << 3);
  v8bf lo = *reinterpret_cast<const v8bf*>(p);
  v8bf hi = *reinterpret_cast<const v8bf*>(p + 16);
  return __builtin_shufflevector(lo, hi, 0, 1, 2, 3, 4, 5, 6, 7,
                                 8, 9, 10, 11, 12, 13, 14, 15);
}

// B-fragment: lane holds N = n0 + (lane&15); element e -> K = k0 + (lane>>4)*8 + (e<8?e:e+8)
DEV_INLINE v16bf load_b32x16(const __bf16* __restrict__ base, int ld, int k0,
                             int n0, int lane) {
  v16bf r;
  const __bf16* p = base + (size_t)(k0 + ((lane >> 4) << 3)) * ld + n0 + (lane & 15);
#pragma unroll
  for (int e = 0; e < 16; ++e) r[e] = p[(size_t)(e < 8 ? e : e + 8) * ld];
  return r;
}

// ---------------- weight convert + transpose: W[O,K] f32 -> Wt[K,O] bf16 ----------------
__global__ void wcast_transpose_kernel(const float* __restrict__ W,
                                       __bf16* __restrict__ Wt, int O, int K) {
  int i = blockIdx.x * blockDim.x + threadIdx.x;
  if (i >= O * K) return;
  int o = i / K, k = i - o * K;
  Wt[(size_t)k * O + o] = (__bf16)W[i];
}

// ---------------- layernorm: one wave per 384-wide row ----------------
template <bool FROM_BF16>
__global__ __launch_bounds__(256) void layernorm_kernel(
    const float* __restrict__ xf, const __bf16* __restrict__ xb,
    const float* __restrict__ g, const float* __restrict__ bta,
    __bf16* __restrict__ out) {
  const int lane = threadIdx.x & 31;
  const int wave = threadIdx.x >> 5;
  const int row = blockIdx.x * 8 + wave;
  const size_t base = (size_t)row * C_;
  float vals[12];
  float s = 0.f;
#pragma unroll
  for (int i = 0; i < 12; ++i) {
    int idx = lane + i * 32;
    float xv = FROM_BF16 ? (float)xb[base + idx] : xf[base + idx];
    vals[i] = xv;
    s += xv;
  }
#pragma unroll
  for (int msk = 16; msk >= 1; msk >>= 1) s += __shfl_xor(s, msk, 32);
  float mu = s * (1.0f / C_);
  float vs = 0.f;
#pragma unroll
  for (int i = 0; i < 12; ++i) {
    float d = vals[i] - mu;
    vs += d * d;
  }
#pragma unroll
  for (int msk = 16; msk >= 1; msk >>= 1) vs += __shfl_xor(vs, msk, 32);
  float rstd = rsqrtf(vs * (1.0f / C_) + EPS_);
#pragma unroll
  for (int i = 0; i < 12; ++i) {
    int idx = lane + i * 32;
    float o = (vals[i] - mu) * rstd * g[idx] + bta[idx];
    out[base + idx] = (__bf16)o;
  }
}

// ---------------- generic WMMA GEMM: out[M,Nn] = A[M,K] * Wt[K,Nn] (+epilogue) --------
// block = 8 waves (4M x 2N), block tile 128x128, wave tile 32x64
enum { MODE_QKV = 0, MODE_PROJ = 1, MODE_FC1 = 2, MODE_FC2 = 3 };

template <int MODE>
__global__ __launch_bounds__(256) void gemm_wmma_kernel(
    const __bf16* __restrict__ A, const __bf16* __restrict__ Wt,
    const float* __restrict__ bias, const float* __restrict__ resid_f32,
    const __bf16* __restrict__ resid_bf16, __bf16* __restrict__ outq,
    __bf16* __restrict__ outkT, __bf16* __restrict__ outv,
    __bf16* __restrict__ out_bf16, float* __restrict__ out_f32, int K, int Nn) {
  const int lane = threadIdx.x & 31;
  const int wave = threadIdx.x >> 5;
  const int wm = wave >> 1, wn = wave & 1;       // 4 (M) x 2 (N) waves
  const int m0 = blockIdx.y * 128 + wm * 32;     // block tile 128 x 128
  const int n0 = blockIdx.x * 128 + wn * 64;     // wave tile  32 x 64

  v8f acc[2][4] = {};
#pragma unroll 2
  for (int k0 = 0; k0 < K; k0 += 32) {
    v16bf a0 = load_a16x32(A, K, m0 + (lane & 15), k0, lane);
    v16bf a1 = load_a16x32(A, K, m0 + 16 + (lane & 15), k0, lane);
    v16bf b0 = load_b32x16(Wt, Nn, k0, n0, lane);
    v16bf b1 = load_b32x16(Wt, Nn, k0, n0 + 16, lane);
    v16bf b2 = load_b32x16(Wt, Nn, k0, n0 + 32, lane);
    v16bf b3 = load_b32x16(Wt, Nn, k0, n0 + 48, lane);
    acc[0][0] = wmma_bf16(a0, b0, acc[0][0]);
    acc[0][1] = wmma_bf16(a0, b1, acc[0][1]);
    acc[0][2] = wmma_bf16(a0, b2, acc[0][2]);
    acc[0][3] = wmma_bf16(a0, b3, acc[0][3]);
    acc[1][0] = wmma_bf16(a1, b0, acc[1][0]);
    acc[1][1] = wmma_bf16(a1, b1, acc[1][1]);
    acc[1][2] = wmma_bf16(a1, b2, acc[1][2]);
    acc[1][3] = wmma_bf16(a1, b3, acc[1][3]);
  }

  const int half = lane >> 4;   // C/D frag: lane -> N=(lane&15), VGPR r -> M=r+8*half
  const int ncol = lane & 15;
#pragma unroll
  for (int mi = 0; mi < 2; ++mi) {
#pragma unroll
    for (int ni = 0; ni < 4; ++ni) {
      const int nb = n0 + ni * 16 + ncol;
      float bval = 0.f;
      if constexpr (MODE != MODE_QKV) bval = bias[nb];
#pragma unroll
      for (int r = 0; r < 8; ++r) {
        const int m = m0 + mi * 16 + r + 8 * half;
        const float vfull = acc[mi][ni][r];
        if constexpr (MODE == MODE_QKV) {
          __bf16 bv = (__bf16)vfull;
          const int b_idx = m >> 11;            // / SEQ
          const int tok = m & (SEQ - 1);
          const int o = nb;
          if (o < C_) {
            int hh = o >> 6, dd = o & 63;
            outq[(((size_t)(b_idx * H_ + hh)) * SEQ + tok) * D_ + dd] = bv;
          } else if (o < 2 * C_) {
            int oo = o - C_;
            int hh = oo >> 6, dd = oo & 63;
            outkT[(((size_t)(b_idx * H_ + hh)) * D_ + dd) * SEQ + tok] = bv;
          } else {
            int oo = o - 2 * C_;
            int hh = oo >> 6, dd = oo & 63;
            outv[(((size_t)(b_idx * H_ + hh)) * SEQ + tok) * D_ + dd] = bv;
          }
        } else if constexpr (MODE == MODE_PROJ) {
          float s = resid_f32[(size_t)m * C_ + nb] + vfull + bval;
          out_bf16[(size_t)m * C_ + nb] = (__bf16)s;   // bf16 residual round
        } else if constexpr (MODE == MODE_FC1) {
          float s = vfull + bval;
          s = 0.5f * s * (1.0f + erff(s * 0.70710678118654752f));  // exact GELU
          out_bf16[(size_t)m * HID_ + nb] = (__bf16)s;
        } else {  // MODE_FC2: final residual, bf16 round, fp32 output
          float s = vfull + bval + (float)resid_bf16[(size_t)m * C_ + nb];
          __bf16 rb = (__bf16)s;
          out_f32[(size_t)m * C_ + nb] = (float)rb;
        }
      }
    }
  }
}

// ---------------- flash attention: 1 wave = 16 query rows, 64-key tiles ----------------
// Per tile: 8 WMMA (S) + 8 WMMA (P.V); only the row-max uses in-loop lane
// reductions (4 ds_bpermute per row per 64 keys); the softmax denominator is
// kept as per-lane partials (rescaled by alpha) and reduced once at the end.
__global__ __launch_bounds__(128) void attn_kernel(const __bf16* __restrict__ q,
                                                   const __bf16* __restrict__ kT,
                                                   const __bf16* __restrict__ v,
                                                   __bf16* __restrict__ o) {
  __shared__ __bf16 smem[4 * 16 * 64];   // wave-private P staging (8 KiB)
  const int lane = threadIdx.x & 31;
  const int wave = threadIdx.x >> 5;
  const int bh = blockIdx.x;             // 0 .. B*H-1
  const int qt = blockIdx.y;             // 0 .. SEQ/64-1
  const int b_idx = bh / H_, hh = bh % H_;
  const int q0 = qt * 64 + wave * 16;

  const __bf16* Qh = q + (size_t)bh * SEQ * D_;
  const __bf16* Kh = kT + (size_t)bh * D_ * SEQ;  // [D, N] (pre-transposed)
  const __bf16* Vh = v + (size_t)bh * SEQ * D_;
  __bf16* pbuf = smem + wave * (16 * 64);

  // Q fragments (resident across the key loop), D=64 split into two K=32 steps
  v16bf aq0 = load_a16x32(Qh, D_, q0 + (lane & 15), 0, lane);
  v16bf aq1 = load_a16x32(Qh, D_, q0 + (lane & 15), 32, lane);

  float mrow[8], lpart[8];
#pragma unroll
  for (int r = 0; r < 8; ++r) { mrow[r] = -3.0e38f; lpart[r] = 0.f; }
  v8f oacc[4] = {};

  const int half = lane >> 4;
  const int nc15 = lane & 15;
  const __bf16* pread = pbuf + nc15 * 64 + (half << 3);

  for (int t = 0; t < SEQ / 64; ++t) {
    const int key0 = t * 64;
    // ---- S = Q K^T over a 16x64 score tile (8 WMMAs) ----
    v8f s[4];
#pragma unroll
    for (int nc = 0; nc < 4; ++nc) {
      v16bf bk0 = load_b32x16(Kh, SEQ, 0, key0 + nc * 16, lane);
      v16bf bk1 = load_b32x16(Kh, SEQ, 32, key0 + nc * 16, lane);
      v8f z = {};
      z = wmma_bf16(aq0, bk0, z);
      s[nc] = wmma_bf16(aq1, bk1, z);
    }

    // ---- online softmax (row max via 16-lane butterfly; sums stay per-lane) ----
    float p[4][8];
#pragma unroll
    for (int r = 0; r < 8; ++r) {
      float a0 = s[0][r] * SCALE, a1 = s[1][r] * SCALE;
      float a2 = s[2][r] * SCALE, a3 = s[3][r] * SCALE;
      float vmax = fmaxf(fmaxf(a0, a1), fmaxf(a2, a3));
      vmax = fmaxf(vmax, __shfl_xor(vmax, 1, 32));
      vmax = fmaxf(vmax, __shfl_xor(vmax, 2, 32));
      vmax = fmaxf(vmax, __shfl_xor(vmax, 4, 32));
      vmax = fmaxf(vmax, __shfl_xor(vmax, 8, 32));
      float mnew = fmaxf(mrow[r], vmax);
      float alpha = __expf(mrow[r] - mnew);
      mrow[r] = mnew;
      float e0 = __expf(a0 - mnew), e1 = __expf(a1 - mnew);
      float e2 = __expf(a2 - mnew), e3 = __expf(a3 - mnew);
      p[0][r] = e0; p[1][r] = e1; p[2][r] = e2; p[3][r] = e3;
      lpart[r] = lpart[r] * alpha + (e0 + e1) + (e2 + e3);
#pragma unroll
      for (int dc = 0; dc < 4; ++dc) oacc[dc][r] *= alpha;
    }

    // ---- C-layout -> A-layout re-shape of P through wave-private LDS ----
#pragma unroll
    for (int r = 0; r < 8; ++r) {
      const int rr = r + 8 * half;
#pragma unroll
      for (int nc = 0; nc < 4; ++nc)
        pbuf[rr * 64 + nc * 16 + nc15] = (__bf16)p[nc][r];
    }

    // ---- O += P V  (two K=32 steps x four 16-wide d chunks = 8 WMMAs) ----
#pragma unroll
    for (int kt = 0; kt < 2; ++kt) {
      const __bf16* pa = pread + kt * 32;
      v8bf lo = *reinterpret_cast<const v8bf*>(pa);
      v8bf hi = *reinterpret_cast<const v8bf*>(pa + 16);
      v16bf pfrag = __builtin_shufflevector(lo, hi, 0, 1, 2, 3, 4, 5, 6, 7,
                                            8, 9, 10, 11, 12, 13, 14, 15);
      const __bf16* vrow = Vh + (size_t)(key0 + kt * 32 + (half << 3)) * D_ + nc15;
#pragma unroll
      for (int dc = 0; dc < 4; ++dc) {
        v16bf bv16;
#pragma unroll
        for (int e = 0; e < 16; ++e) bv16[e] = vrow[(e < 8 ? e : e + 8) * D_ + dc * 16];
        oacc[dc] = wmma_bf16(pfrag, bv16, oacc[dc]);
      }
    }
  }

  // ---- final denominator reduction (once), normalize, scatter to [B,N,H*D] ----
#pragma unroll
  for (int r = 0; r < 8; ++r) {
    float l = lpart[r];
    l += __shfl_xor(l, 1, 32);
    l += __shfl_xor(l, 2, 32);
    l += __shfl_xor(l, 4, 32);
    l += __shfl_xor(l, 8, 32);
    float inv = 1.0f / l;
    const int qr = q0 + r + 8 * half;
    const size_t outbase = ((size_t)(b_idx * SEQ + qr)) * C_ + hh * D_;
#pragma unroll
    for (int dc = 0; dc < 4; ++dc)
      o[outbase + dc * 16 + nc15] = (__bf16)(oacc[dc][r] * inv);
  }
}

// ---------------- host orchestration ----------------
extern "C" void kernel_launch(void* const* d_in, const int* in_sizes, int n_in,
                              void* d_out, int out_size, void* d_ws,
                              size_t ws_size, hipStream_t stream) {
  (void)in_sizes; (void)n_in; (void)out_size; (void)ws_size;
  const float* x = (const float*)d_in[0];
  const float* ln1_g = (const float*)d_in[1];
  const float* ln1_b = (const float*)d_in[2];
  const float* w_qkv = (const float*)d_in[3];
  const float* w_proj = (const float*)d_in[4];
  const float* b_proj = (const float*)d_in[5];
  const float* ln2_g = (const float*)d_in[6];
  const float* ln2_b = (const float*)d_in[7];
  const float* w_fc1 = (const float*)d_in[8];
  const float* b_fc1 = (const float*)d_in[9];
  const float* w_fc2 = (const float*)d_in[10];
  const float* b_fc2 = (const float*)d_in[11];
  float* out = (float*)d_out;

  char* ws = (char*)d_ws;
  __bf16* wqkv_t = (__bf16*)(ws + OFF_WQKV);
  __bf16* wproj_t = (__bf16*)(ws + OFF_WPROJ);
  __bf16* wfc1_t = (__bf16*)(ws + OFF_WFC1);
  __bf16* wfc2_t = (__bf16*)(ws + OFF_WFC2);
  __bf16* h_bf = (__bf16*)(ws + OFF_H);
  __bf16* qbuf = (__bf16*)(ws + OFF_Q);
  __bf16* ktbuf = (__bf16*)(ws + OFF_KT);
  __bf16* vbuf = (__bf16*)(ws + OFF_V);
  __bf16* obuf = (__bf16*)(ws + OFF_O);
  __bf16* x1buf = (__bf16*)(ws + OFF_X1);
  __bf16* hidbuf = (__bf16*)(ws + OFF_HID);

  // 0) weights -> bf16, transposed to [K, N] for direct B-fragment loads
  wcast_transpose_kernel<<<(3 * C_ * C_ + 255) / 256, 256, 0, stream>>>(
      w_qkv, wqkv_t, 3 * C_, C_);
  wcast_transpose_kernel<<<(C_ * C_ + 255) / 256, 256, 0, stream>>>(
      w_proj, wproj_t, C_, C_);
  wcast_transpose_kernel<<<(HID_ * C_ + 255) / 256, 256, 0, stream>>>(
      w_fc1, wfc1_t, HID_, C_);
  wcast_transpose_kernel<<<(C_ * HID_ + 255) / 256, 256, 0, stream>>>(
      w_fc2, wfc2_t, C_, HID_);

  // 1) LN1
  layernorm_kernel<false><<<M_TOK / 8, 256, 0, stream>>>(x, nullptr, ln1_g,
                                                         ln1_b, h_bf);
  // 2) QKV GEMM (scatters q / kT / v)
  {
    dim3 g(3 * C_ / 128, M_TOK / 128);
    gemm_wmma_kernel<MODE_QKV><<<g, 256, 0, stream>>>(
        h_bf, wqkv_t, nullptr, nullptr, nullptr, qbuf, ktbuf, vbuf, nullptr,
        nullptr, C_, 3 * C_);
  }
  // 3) attention
  {
    dim3 g(B_ * H_, SEQ / 64);
    attn_kernel<<<g, 128, 0, stream>>>(qbuf, ktbuf, vbuf, obuf);
  }
  // 4) proj + residual -> x1 (bf16-rounded)
  {
    dim3 g(C_ / 128, M_TOK / 128);
    gemm_wmma_kernel<MODE_PROJ><<<g, 256, 0, stream>>>(
        obuf, wproj_t, b_proj, x, nullptr, nullptr, nullptr, nullptr, x1buf,
        nullptr, C_, C_);
  }
  // 5) LN2
  layernorm_kernel<true><<<M_TOK / 8, 256, 0, stream>>>(nullptr, x1buf, ln2_g,
                                                        ln2_b, h_bf);
  // 6) FC1 + GELU
  {
    dim3 g(HID_ / 128, M_TOK / 128);
    gemm_wmma_kernel<MODE_FC1><<<g, 256, 0, stream>>>(
        h_bf, wfc1_t, b_fc1, nullptr, nullptr, nullptr, nullptr, nullptr,
        hidbuf, nullptr, C_, HID_);
  }
  // 7) FC2 + residual -> fp32 output (bf16-rounded values)
  {
    dim3 g(C_ / 128, M_TOK / 128);
    gemm_wmma_kernel<MODE_FC2><<<g, 256, 0, stream>>>(
        hidbuf, wfc2_t, b_fc2, nullptr, x1buf, nullptr, nullptr, nullptr,
        nullptr, out, HID_, C_);
  }
}
